// Head_38895223833068
// MI455X (gfx1250) — matrix operand order repaired and checked by reading, not compiled, not run on hardware
//
#include <hip/hip_runtime.h>
#include <hip/hip_bf16.h>

// Problem constants (from the reference)
#define EMB   1024
#define HD    64
#define SLEN  4096
#define BATCH 4

typedef __attribute__((ext_vector_type(16))) _Float16 v16h;
typedef __attribute__((ext_vector_type(8)))  _Float16 v8h;
typedef __attribute__((ext_vector_type(8)))  float    v8f;

// Keep fragment loads above the WMMA burst (no instruction may cross).
#define SCHED_FENCE() __builtin_amdgcn_sched_barrier(0)

// ---------------------------------------------------------------------------
// Kernel 1: QKV projection.  out[row][n] = sum_k x[row][k] * W[n][k]
// Tile: 128 rows x 64 cols per workgroup (8 waves, each wave = 16 rows x 64
// cols = 4 WMMA 16x16 accumulators).  K-loop over EMB in steps of 32.
// grid.x = (B*S)/128 row tiles, grid.y = {0,1,2} selects Wq/Wk/Wv.
// ---------------------------------------------------------------------------
__global__ __launch_bounds__(256) void qkv_proj_kernel(
    const float* __restrict__ x,
    const float* __restrict__ Wq,
    const float* __restrict__ Wk,
    const float* __restrict__ Wv,
    _Float16* __restrict__ Qh,
    _Float16* __restrict__ Kh,
    _Float16* __restrict__ Vh)
{
    __shared__ _Float16 Xs[128][32];   // x tile (f16)       8 KB
    __shared__ _Float16 Ws[64][32];    // W tile (f16)       4 KB

    const float* W   = (blockIdx.y == 0) ? Wq : (blockIdx.y == 1) ? Wk : Wv;
    _Float16*    Out = (blockIdx.y == 0) ? Qh : (blockIdx.y == 1) ? Kh : Vh;

    const int t    = threadIdx.x;      // 0..255  (8 wave32 waves)
    const int wave = t >> 5;
    const int lane = t & 31;
    const int half = lane >> 4;        // 0: lanes 0-15, 1: lanes 16-31
    const int ln   = lane & 15;
    const long rowBase = (long)blockIdx.x * 128;

    v8f acc[4] = {};                   // 16x64 fp32 accumulators per wave

    for (int k0 = 0; k0 < EMB; k0 += 32) {
        __syncthreads();
        // ---- stage x tile: 128x32 fp32 -> f16, coalesced float4 loads ----
        {
            const int i0 = t * 16;            // 16 elements per thread
            const int r  = i0 >> 5;           // row in tile
            const int c  = i0 & 31;           // col in tile (0 or 16)
            const float4* src = (const float4*)(x + (rowBase + r) * EMB + k0 + c);
#pragma unroll
            for (int j = 0; j < 4; ++j) {
                float4 v = src[j];
                int cc = c + j * 4;
                Xs[r][cc + 0] = (_Float16)v.x;
                Xs[r][cc + 1] = (_Float16)v.y;
                Xs[r][cc + 2] = (_Float16)v.z;
                Xs[r][cc + 3] = (_Float16)v.w;
            }
            // prefetch next x K-chunk (emits global_prefetch_b8)
            if (k0 + 32 < EMB)
                __builtin_prefetch(x + (rowBase + r) * EMB + k0 + 32 + c, 0, 1);
        }
        // ---- stage W tile: 64x32 fp32 -> f16 ----
        {
            const int r = t >> 2;             // output col n (0..63)
            const int c = (t & 3) * 8;
            const float4* src = (const float4*)(W + r * EMB + k0 + c);
#pragma unroll
            for (int j = 0; j < 2; ++j) {
                float4 v = src[j];
                int cc = c + j * 4;
                Ws[r][cc + 0] = (_Float16)v.x;
                Ws[r][cc + 1] = (_Float16)v.y;
                Ws[r][cc + 2] = (_Float16)v.z;
                Ws[r][cc + 3] = (_Float16)v.w;
            }
        }
        __syncthreads();

        // ---- A fragment (16x32 f16): lane = M, K = half*16+e ----
        v16h a;
#pragma unroll
        for (int e = 0; e < 8; ++e)
            a[e] = Xs[wave * 16 + ln][half * 16 + e];
#pragma unroll
        for (int e = 0; e < 8; ++e)
            a[8 + e] = Xs[wave * 16 + ln][half * 16 + 8 + e];

        // ---- load ALL B fragments, fence the scheduler, then issue the four
        //      WMMAs back-to-back (one dscnt wait instead of four) ----
        v16h b[4];
#pragma unroll
        for (int nt = 0; nt < 4; ++nt)
#pragma unroll
            for (int e = 0; e < 16; ++e)
                b[nt][e] = Ws[nt * 16 + ln][half * 16 + e];
        SCHED_FENCE();
#pragma unroll
        for (int nt = 0; nt < 4; ++nt)
            acc[nt] = __builtin_amdgcn_wmma_f32_16x16x32_f16(
                false, a, false, b[nt], (short)0, acc[nt], false, false);
    }

    // ---- store f16 Q/K/V: C layout VGPR r -> row r + 8*half, col = lane&15 ----
#pragma unroll
    for (int nt = 0; nt < 4; ++nt)
#pragma unroll
        for (int r = 0; r < 8; ++r) {
            long row = rowBase + wave * 16 + r + 8 * half;
            Out[row * HD + nt * 16 + ln] = (_Float16)acc[nt][r];
        }
}

// ---------------------------------------------------------------------------
// Kernel 2: causal flash attention.
// grid.x = S/128 query blocks, grid.y = batch.  8 waves; wave w owns queries
// [qblk*128 + w*16, +16).  KV blocks of 64, online softmax in fp32 (base-2).
// V is staged *transposed* (Vt[h][kv]) so both B-fragment reads are
// contiguous ds_load_b128; rows padded to 72 halves for bank spread.
// ---------------------------------------------------------------------------
#define PAD 72

__global__ __launch_bounds__(256) void flash_attn_kernel(
    const _Float16* __restrict__ Qh,
    const _Float16* __restrict__ Kh,
    const _Float16* __restrict__ Vh,
    float* __restrict__ out)
{
    __shared__ _Float16 Ks[64][PAD];        // [kv][h]   9.2 KB
    __shared__ _Float16 Vt[64][PAD];        // [h][kv]   9.2 KB (transposed)
    __shared__ _Float16 Ps[8][16][PAD];     // per-wave P tiles [wave][q][kv] 18.4 KB

    const int t    = threadIdx.x;
    const int wave = t >> 5;
    const int lane = t & 31;
    const int half = lane >> 4;
    const int ln   = lane & 15;
    const int b    = blockIdx.y;
    const int qblk = blockIdx.x;
    const long base  = (long)b * SLEN;      // batch row offset
    const int  qBase = qblk * 128 + wave * 16;

    // ---- Q fragments for this wave: 16 q x 64 h -> two 16x32 A-frags ----
    v16h qa[2];
#pragma unroll
    for (int s = 0; s < 2; ++s) {
        const _Float16* qrow = Qh + (base + qBase + ln) * HD + s * 32;
#pragma unroll
        for (int e = 0; e < 8; ++e) qa[s][e]     = qrow[half * 16 + e];
#pragma unroll
        for (int e = 0; e < 8; ++e) qa[s][8 + e] = qrow[half * 16 + 8 + e];
    }

    v8f acc[4] = {};                        // O accumulator, 16 q x 64 h
    float mrow[8], lrow[8];
#pragma unroll
    for (int r = 0; r < 8; ++r) { mrow[r] = -1e30f; lrow[r] = 0.0f; }

    // scores in base-2 domain: p = exp2(s*scale2 - m2); v_exp_f32 is natively 2^x
    const float scale2 = 0.03125f * 1.44269504f;   // (1/sqrt(EMB)) * log2(e)
    const int kvEnd = qblk * 128 + 128;            // causal bound for this WG

    for (int kv0 = 0; kv0 < kvEnd; kv0 += 64) {
        __syncthreads();
        // ---- stage K (row-major) and V (transposed) blocks, 64x64 f16 ----
        {
            const int r = t >> 2;            // kv row 0..63
            const int c = (t & 3) * 16;      // h col 0,16,32,48
            const v8h* ksrc = (const v8h*)(Kh + (base + kv0 + r) * HD + c);
            const v8h* vsrc = (const v8h*)(Vh + (base + kv0 + r) * HD + c);
            v8h k0 = ksrc[0], k1 = ksrc[1];
            v8h w0 = vsrc[0], w1 = vsrc[1];
            *(v8h*)&Ks[r][c]     = k0;
            *(v8h*)&Ks[r][c + 8] = k1;
#pragma unroll
            for (int j = 0; j < 8; ++j) Vt[c + j][r]     = w0[j];
#pragma unroll
            for (int j = 0; j < 8; ++j) Vt[c + 8 + j][r] = w1[j];
            // prefetch next KV block (global_prefetch_b8)
            if (kv0 + 64 < kvEnd) {
                __builtin_prefetch(Kh + (base + kv0 + 64 + r) * HD + c, 0, 1);
                __builtin_prefetch(Vh + (base + kv0 + 64 + r) * HD + c, 0, 1);
            }
        }
        __syncthreads();

        // Wave-uniform skip of KV blocks entirely above this wave's diagonal
        // (branch is uniform per wave -> EXEC stays all-ones for WMMA).
        if (kv0 <= qBase + 15) {
            // ---- scores: S = Q (16x64) x K^T (64x64): 2 k-steps x 4 n-tiles.
            //      Load 4 B-frags, fence, then 4 back-to-back WMMAs ----
            v8f sc[4] = {};
#pragma unroll
            for (int s = 0; s < 2; ++s) {
                v16h bf[4];                  // B[k=h][n=kv] = Ks[kv][h]
#pragma unroll
                for (int nt = 0; nt < 4; ++nt)
#pragma unroll
                    for (int e = 0; e < 16; ++e)
                        bf[nt][e] = Ks[nt * 16 + ln][s * 32 + half * 16 + e];
                SCHED_FENCE();
#pragma unroll
                for (int nt = 0; nt < 4; ++nt)
                    sc[nt] = __builtin_amdgcn_wmma_f32_16x16x32_f16(
                        false, qa[s], false, bf[nt], (short)0, sc[nt], false, false);
            }

            // ---- scale (+ causal mask only on diagonal blocks) ----
            if (kv0 + 63 <= qBase) {         // fully visible: no mask needed
#pragma unroll
                for (int nt = 0; nt < 4; ++nt)
#pragma unroll
                    for (int r = 0; r < 8; ++r)
                        sc[nt][r] *= scale2;
            } else {
#pragma unroll
                for (int nt = 0; nt < 4; ++nt) {
                    const int kvcol = kv0 + nt * 16 + ln;
#pragma unroll
                    for (int r = 0; r < 8; ++r) {
                        const int qrow = qBase + r + 8 * half;
                        float v = sc[nt][r] * scale2;
                        sc[nt][r] = (kvcol <= qrow) ? v : -1e30f;
                    }
                }
            }

            // ---- online softmax: row max via shuffles over 16-lane half-group ----
            float mnew[8];
#pragma unroll
            for (int r = 0; r < 8; ++r) {
                float v = fmaxf(fmaxf(sc[0][r], sc[1][r]), fmaxf(sc[2][r], sc[3][r]));
#pragma unroll
                for (int off = 8; off >= 1; off >>= 1)
                    v = fmaxf(v, __shfl_xor(v, off, 32));
                mnew[r] = fmaxf(mrow[r], v);
            }
#pragma unroll
            for (int r = 0; r < 8; ++r) {
                const float corr = exp2f(mrow[r] - mnew[r]);
                lrow[r] *= corr;
#pragma unroll
                for (int nt = 0; nt < 4; ++nt) acc[nt][r] *= corr;
                mrow[r] = mnew[r];
            }
            float lpart[8];
#pragma unroll
            for (int r = 0; r < 8; ++r) lpart[r] = 0.0f;
#pragma unroll
            for (int nt = 0; nt < 4; ++nt)
#pragma unroll
                for (int r = 0; r < 8; ++r) {
                    const float p = exp2f(sc[nt][r] - mrow[r]);
                    sc[nt][r] = p;
                    lpart[r] += p;
                }
#pragma unroll
            for (int r = 0; r < 8; ++r) {
                float v = lpart[r];
#pragma unroll
                for (int off = 8; off >= 1; off >>= 1)
                    v += __shfl_xor(v, off, 32);
                lrow[r] += v;
            }

            // ---- C-layout P -> A-layout via per-wave LDS tile (same-wave LDS
            //      ops are in-order per DScnt rules: no barrier needed) ----
#pragma unroll
            for (int nt = 0; nt < 4; ++nt)
#pragma unroll
                for (int r = 0; r < 8; ++r)
                    Ps[wave][r + 8 * half][nt * 16 + ln] = (_Float16)sc[nt][r];

            // ---- O += P (16x64) x V (64x64); batch frag loads, fence, WMMAs ----
#pragma unroll
            for (int s = 0; s < 2; ++s) {
                v16h pa;
#pragma unroll
                for (int e = 0; e < 8; ++e)
                    pa[e] = Ps[wave][ln][s * 32 + half * 16 + e];
#pragma unroll
                for (int e = 0; e < 8; ++e)
                    pa[8 + e] = Ps[wave][ln][s * 32 + half * 16 + 8 + e];
                v16h bf[4];                  // B[k=kv][n=h] = Vt[h][kv]
#pragma unroll
                for (int nt = 0; nt < 4; ++nt)
#pragma unroll
                    for (int e = 0; e < 16; ++e)
                        bf[nt][e] = Vt[nt * 16 + ln][s * 32 + half * 16 + e];
                SCHED_FENCE();
#pragma unroll
                for (int nt = 0; nt < 4; ++nt)
                    acc[nt] = __builtin_amdgcn_wmma_f32_16x16x32_f16(
                        false, pa, false, bf[nt], (short)0, acc[nt], false, false);
            }
        }
    }

    // ---- epilogue: normalize and store fp32 output [B,S,H] ----
#pragma unroll
    for (int nt = 0; nt < 4; ++nt)
#pragma unroll
        for (int r = 0; r < 8; ++r) {
            const long row = base + qBase + r + 8 * half;
            out[row * HD + nt * 16 + ln] = acc[nt][r] / lrow[r];
        }
}

// ---------------------------------------------------------------------------
extern "C" void kernel_launch(void* const* d_in, const int* in_sizes, int n_in,
                              void* d_out, int out_size, void* d_ws, size_t ws_size,
                              hipStream_t stream)
{
    const float* x  = (const float*)d_in[0];
    const float* Wq = (const float*)d_in[1];
    const float* Wk = (const float*)d_in[2];
    const float* Wv = (const float*)d_in[3];

    // Workspace: f16 Q, K, V  (each B*S*HD = 1M elements = 2 MB)
    const size_t qkvElems = (size_t)BATCH * SLEN * HD;
    _Float16* Qh = (_Float16*)d_ws;
    _Float16* Kh = Qh + qkvElems;
    _Float16* Vh = Kh + qkvElems;

    dim3 gProj((BATCH * SLEN) / 128, 3);
    qkv_proj_kernel<<<gProj, 256, 0, stream>>>(x, Wq, Wk, Wv, Qh, Kh, Vh);

    dim3 gAttn(SLEN / 128, BATCH);
    flash_attn_kernel<<<gAttn, 256, 0, stream>>>(Qh, Kh, Vh, (float*)d_out);
}
